// PWD_LAYER_32804960207252
// MI455X (gfx1250) — matrix operand order repaired and checked by reading, not compiled, not run on hardware
//
#include <hip/hip_runtime.h>
#include <cstddef>

// ---------------------------------------------------------------------------
// Problem constants (fixed by the reference): B=8, C=128, H=W=48, N=2304.
// P[b,i,j] = sx[b,i] + sy[b,j] - 2 * (y_i . x_j), batched GEMM K=128.
// ---------------------------------------------------------------------------
constexpr int B_ = 8;
constexpr int C_ = 128;
constexpr int H_ = 48;
constexpr int W_ = 48;
constexpr int N_ = H_ * W_;            // 2304
constexpr int PACK_NT = N_ / 32;       // 72 pack tiles of 32 pixels

constexpr int TILE = 64;               // 64x64 output tile per wave
constexpr int NT = N_ / TILE;          // 36
constexpr int TILES_PER_B = NT * NT;   // 1296
constexpr int WAVES_PER_BLOCK = 8;     // 256 threads, wave32

typedef _Float16 v8h  __attribute__((ext_vector_type(8)));
typedef _Float16 v16h __attribute__((ext_vector_type(16)));
typedef _Float16 h2   __attribute__((ext_vector_type(2)));
typedef float    v8f  __attribute__((ext_vector_type(8)));

// ---------------------------------------------------------------------------
// Kernel 1: repack (B,C,N) f32 -> (B,N,C) f16 (transposed, so each pixel's
// 128-channel vector is contiguous) and compute per-pixel squared norms in
// f32 from the original data. One block handles 32 pixels x all 128 channels.
// Global reads are coalesced along N; writes are coalesced along C via LDS.
// ---------------------------------------------------------------------------
__global__ __launch_bounds__(256) void pwd_pack_kernel(
    const float* __restrict__ src,   // (B, C, N) f32
    _Float16* __restrict__ dsth,     // (B, N, C) f16
    float* __restrict__ dstn)        // (B, N)    f32 squared norms
{
    __shared__ float tile[C_][33];   // padded to dodge bank conflicts
    __shared__ float part[8][32];

    const int b  = blockIdx.x / PACK_NT;
    const int n0 = (blockIdx.x - b * PACK_NT) * 32;
    const int t  = threadIdx.x;

    // Load phase: thread t reads channels [c0, c0+16) of pixel n0+n.
    const int n  = t & 31;
    const int c0 = (t >> 5) * 16;
    const float* sp = src + (size_t)b * C_ * N_ + n0 + n;
    float nrm = 0.0f;
#pragma unroll
    for (int k = 0; k < 16; ++k) {
        float v = sp[(size_t)(c0 + k) * N_];
        tile[c0 + k][n] = v;
        nrm = fmaf(v, v, nrm);
    }
    part[t >> 5][n] = nrm;
    __syncthreads();

    // Write phase: transpose out of LDS, 2 f16 per 32-bit store, contiguous
    // along C across adjacent lanes -> coalesced.
    const int c2 = t & 63;           // packed pair index: channels 2*c2, 2*c2+1
    const int nb = t >> 6;           // 0..3
    h2* outp = (h2*)dsth;
#pragma unroll
    for (int p = 0; p < 8; ++p) {
        int nn = nb + p * 4;         // 0..31
        h2 w;
        w[0] = (_Float16)tile[2 * c2][nn];
        w[1] = (_Float16)tile[2 * c2 + 1][nn];
        outp[((size_t)b * N_ + n0 + nn) * (C_ / 2) + c2] = w;
    }

    if (t < 32) {
        float s = 0.0f;
#pragma unroll
        for (int p = 0; p < 8; ++p) s += part[p][t];
        dstn[(size_t)b * N_ + n0 + t] = s;
    }
}

// ---------------------------------------------------------------------------
// Kernel 2: WMMA GEMM + fused epilogue.
// Each wave computes a 64x64 tile of P[b] via 4x4 accumulator tiles of
// v_wmma_f32_16x16x32_f16, K loop = 4 steps of 32 channels (64 WMMAs/wave).
// 64x64 register tiling keeps L2 operand traffic (~332 MB total) at ~2x the
// 170 MB HBM output-write floor, so the kernel stays write-bandwidth bound.
//
// Fragment layouts per CDNA5 ISA 7.12.2 (wave32):
//   A (16x32 f16): lane L holds row M = L&15; per-lane halves are
//     K = [8*hi .. 8*hi+7]  (VGPRs 0-3) and K = [16+8*hi .. 16+8*hi+7].
//   B (32x16 f16): lane L holds col N = L&15; same split halves as A:
//     K = [8*hi .. 8*hi+7] and K = [16+8*hi .. 16+8*hi+7].
//   C/D (16x16 f32): VGPR r, lane L -> element (M = r + 8*hi, N = L&15).
// ---------------------------------------------------------------------------
__global__ __launch_bounds__(256) void pwd_wmma_kernel(
    const _Float16* __restrict__ yh, // (B, N, C) f16  -> A matrix rows (i)
    const _Float16* __restrict__ xh, // (B, N, C) f16  -> B matrix cols (j)
    const float* __restrict__ sx,    // (B, N) ||x||^2 -> row term (i)
    const float* __restrict__ sy,    // (B, N) ||y||^2 -> col term (j)
    float* __restrict__ out)         // (B, N, N)
{
    const int lane = threadIdx.x & 31;
    const int wave = threadIdx.x >> 5;
    const int gw   = blockIdx.x * WAVES_PER_BLOCK + wave;

    const int b  = gw / TILES_PER_B;
    const int r  = gw - b * TILES_PER_B;
    const int ti = r / NT;
    const int tj = r - ti * NT;
    const int i0 = ti * TILE;
    const int j0 = tj * TILE;

    const int l16 = lane & 15;
    const int hi  = lane >> 4;

    const _Float16* arow = yh + ((size_t)b * N_ + i0 + l16) * C_;
    const _Float16* bcol = xh + ((size_t)b * N_ + j0 + l16) * C_;

    v8f acc[4][4] = {};

#pragma unroll
    for (int k0 = 0; k0 < C_; k0 += 32) {
        v16h A[4];
        v16h Bm[4];
#pragma unroll
        for (int it = 0; it < 4; ++it) {
            const _Float16* p = arow + (size_t)(16 * it) * C_;
            // A fragment: split-half per-lane layout
            v8h lo = *(const v8h*)(p + k0 + 8 * hi);
            v8h hi8 = *(const v8h*)(p + k0 + 16 + 8 * hi);
            A[it] = __builtin_shufflevector(lo, hi8,
                        0,1,2,3,4,5,6,7,8,9,10,11,12,13,14,15);
        }
#pragma unroll
        for (int jt = 0; jt < 4; ++jt) {
            const _Float16* p = bcol + (size_t)(16 * jt) * C_;
            // B fragment: split-half per-lane layout (same K order as A)
            v8h lo = *(const v8h*)(p + k0 + 8 * hi);
            v8h hi8 = *(const v8h*)(p + k0 + 16 + 8 * hi);
            Bm[jt] = __builtin_shufflevector(lo, hi8,
                        0,1,2,3,4,5,6,7,8,9,10,11,12,13,14,15);
        }
#pragma unroll
        for (int it = 0; it < 4; ++it) {
#pragma unroll
            for (int jt = 0; jt < 4; ++jt) {
                acc[it][jt] = __builtin_amdgcn_wmma_f32_16x16x32_f16(
                    false, A[it], false, Bm[jt], (short)0, acc[it][jt],
                    false, false);
            }
        }
    }

    // Fused epilogue: P = sx[i] + sy[j] - 2*zz
    float syc[4];
#pragma unroll
    for (int jt = 0; jt < 4; ++jt)
        syc[jt] = sy[(size_t)b * N_ + j0 + jt * 16 + l16];

    float* outb = out + (size_t)b * N_ * N_;

#pragma unroll
    for (int it = 0; it < 4; ++it) {
        const int ibase = i0 + it * 16 + 8 * hi;
        const float* sxr = sx + (size_t)b * N_ + ibase;
        float sxv[8];
#pragma unroll
        for (int q = 0; q < 8; ++q) sxv[q] = sxr[q]; // contiguous 32B load
#pragma unroll
        for (int jt = 0; jt < 4; ++jt) {
            const int j = j0 + jt * 16 + l16;
#pragma unroll
            for (int q = 0; q < 8; ++q) {
                outb[(size_t)(ibase + q) * N_ + j] =
                    fmaf(-2.0f, acc[it][jt][q], sxv[q] + syc[jt]);
            }
        }
    }
}

// ---------------------------------------------------------------------------
// Launch: d_in[0]=x (B,C,H,W) f32, d_in[1]=y. d_out = (B, N, N) f32.
// Workspace: xh | yh | sx | sy  (~9.6 MB).
// ---------------------------------------------------------------------------
extern "C" void kernel_launch(void* const* d_in, const int* in_sizes, int n_in,
                              void* d_out, int out_size, void* d_ws, size_t ws_size,
                              hipStream_t stream) {
    const float* x = (const float*)d_in[0];
    const float* y = (const float*)d_in[1];
    float* out = (float*)d_out;

    char* ws = (char*)d_ws;
    _Float16* xh = (_Float16*)ws;                          // B*N*C f16
    _Float16* yh = xh + (size_t)B_ * N_ * C_;              // B*N*C f16
    float* sx = (float*)(yh + (size_t)B_ * N_ * C_);       // B*N f32
    float* sy = sx + (size_t)B_ * N_;                      // B*N f32

    dim3 blk(256);
    pwd_pack_kernel<<<dim3(B_ * PACK_NT), blk, 0, stream>>>(x, xh, sx);
    pwd_pack_kernel<<<dim3(B_ * PACK_NT), blk, 0, stream>>>(y, yh, sy);

    const int nblocks = (B_ * TILES_PER_B) / WAVES_PER_BLOCK; // 1296
    pwd_wmma_kernel<<<dim3(nblocks), blk, 0, stream>>>(yh, xh, sx, sy, out);
}
